// LSTM_57346403336563
// MI455X (gfx1250) — matrix operand order repaired and checked by reading, not compiled, not run on hardware
//
#include <hip/hip_runtime.h>
#include <cstdint>
#include <cstddef>

#define B_  64
#define T_  512
#define IN_ 128
#define H_  512
#define G4  2048   // 4*H

// scan decomposition: 4 batch groups x 16 column-split workgroups
#define NSPLIT 16
#define WGCOLS 32           // h columns per scan workgroup
#define HEXSTRIDE (16 * H_) // u16 elements per batch group in the h-exchange buffer

typedef __attribute__((ext_vector_type(16))) __bf16         v16bf;
typedef __attribute__((ext_vector_type(16))) unsigned short v16u;
typedef __attribute__((ext_vector_type(8)))  unsigned short v8u;
typedef __attribute__((ext_vector_type(8)))  float          v8f;

__device__ __forceinline__ unsigned short f2bf(float x) {
    unsigned int u = __float_as_uint(x);
    return (unsigned short)((u + 0x7FFFu + ((u >> 16) & 1u)) >> 16);  // RNE
}
__device__ __forceinline__ float sigmoidf_(float x) { return 1.f / (1.f + __expf(-x)); }

__device__ __forceinline__ v16u cat16(v8u lo, v8u hi) {
    v16u r;
#pragma unroll
    for (int i = 0; i < 8; ++i) { r[i] = lo[i]; r[i + 8] = hi[i]; }
    return r;
}

__device__ __forceinline__ v8f wmma_bf16(v16u a, v16u b, v8f c) {
    return __builtin_amdgcn_wmma_f32_16x16x32_bf16(
        false, __builtin_bit_cast(v16bf, a),
        false, __builtin_bit_cast(v16bf, b),
        (short)0, c, false, false);
}

// ---------------------------------------------------------------------------
// Pack row-major f32 weight (N x K), used as B = W^T, into WMMA B-fragment
// order: [ntile][kblock][lane][16 bf16]. Lane l (lr=l%16, half=l/16) holds
// column n = nt*16+lr, K = kb*32 + half*16 + p  (p contiguous -> 32B/lane).
// ---------------------------------------------------------------------------
__global__ void pack_w(const float* __restrict__ W, unsigned short* __restrict__ out,
                       int N, int K) {
    int idx = blockIdx.x * blockDim.x + threadIdx.x;
    if (idx >= N * K) return;
    int p    = idx & 15;
    int l    = (idx >> 4) & 31;
    int rest = idx >> 9;
    int KB   = K >> 5;
    int kb   = rest % KB;
    int nt   = rest / KB;
    int n    = nt * 16 + (l & 15);
    int k    = kb * 32 + (l >> 4) * 16 + p;
    out[idx] = f2bf(W[(size_t)n * K + k]);
}

// ---------------------------------------------------------------------------
// XG = A(f32, M x K) @ Wpacked^T + (b1 + b2), output f32 M x 2048.
// One block = one 16-row M-tile staged to LDS as bf16; 8 waves x 16 N-tiles.
// ---------------------------------------------------------------------------
__global__ void __launch_bounds__(256)
gemm_xg(const float* __restrict__ A, const unsigned short* __restrict__ Bp,
        const float* __restrict__ b1, const float* __restrict__ b2,
        float* __restrict__ Out, int K) {
    __shared__ unsigned short lds_a[16 * 512];
    const int tid = threadIdx.x;
    const int m0  = blockIdx.x * 16;

    for (int idx = tid; idx < 16 * K; idx += 256) {
        int r = idx / K, k = idx - r * K;
        lds_a[idx] = f2bf(A[(size_t)(m0 + r) * K + k]);
    }
    __syncthreads();

    const int wid  = tid >> 5;
    const int lane = tid & 31;
    const int lr   = lane & 15;
    const int half = lane >> 4;
    const int KB   = K >> 5;

    for (int i = 0; i < 16; ++i) {
        const int nt = wid * 16 + i;
        v8f acc = {};
        for (int kb = 0; kb < KB; ++kb) {
            const int k0 = kb * 32;
            v16u af = cat16(*(const v8u*)&lds_a[lr * K + k0 + half * 16],
                            *(const v8u*)&lds_a[lr * K + k0 + half * 16 + 8]);
            const unsigned short* bptr =
                Bp + ((size_t)(nt * KB + kb) * 32 + lane) * 16;
            __builtin_prefetch(bptr + 512, 0, 1);  // next k-block
            v16u bf = cat16(*(const v8u*)bptr, *(const v8u*)(bptr + 8));
            acc = wmma_bf16(af, bf, acc);
        }
        const float bias = b1[nt * 16 + lr] + b2[nt * 16 + lr];
        float* op = Out + (size_t)(m0 + half * 8) * (size_t)G4 + nt * 16 + lr;
#pragma unroll
        for (int v = 0; v < 8; ++v) op[(size_t)v * G4] = acc[v] + bias;
    }
}

// Zero the h-exchange buffer and the inter-workgroup barrier counters.
__global__ void scan_init(unsigned short* __restrict__ Hex, unsigned int* __restrict__ Bar) {
    int tid = blockIdx.x * blockDim.x + threadIdx.x;
    if (tid < 4) Bar[tid] = 0;
    for (int i = tid; i < 4 * HEXSTRIDE; i += gridDim.x * blockDim.x) Hex[i] = 0;
}

// ---------------------------------------------------------------------------
// Recurrent scan with LDS-resident weights.
// Grid = 64 blocks of 256 threads: g = blk>>4 selects 16 batch rows,
// s = blk&15 selects 32 h-columns. Each block keeps its 128 KB W_hh slice in
// LDS for all 512 steps (no per-step L2 weight streaming). Per step:
//   stage h (16KB bf16) from global exchange buffer -> LDS
//   8 waves x 1 gate-tile GEMM (16 x v_wmma_f32_16x16x32_bf16 from LDS)
//   + XG, gates fused via 8KB LDS exchange, c in registers,
//   h -> global exchange, device-scope atomic barrier over the 16 blocks
//   of the batch group, then global_inv so next step can't read stale WGP$.
// ---------------------------------------------------------------------------
__global__ void __launch_bounds__(256)
lstm_scan(const float* __restrict__ XG, const unsigned short* __restrict__ Wp,
          float* __restrict__ HSeq, float* __restrict__ HLast,
          unsigned short* __restrict__ Hex, unsigned int* __restrict__ Bar) {
    __shared__ unsigned short w_lds[8 * 16 * 32 * 16];  // 128 KB: 8 ntiles of B-fragments
    __shared__ unsigned short h_lds[16 * H_];           // 16 KB: h (bf16, row-major)
    __shared__ float          g_lds[4][16][WGCOLS];     // 8 KB: gate pre-activations

    const int tid  = threadIdx.x;
    const int g    = blockIdx.x >> 4;   // batch group (16 rows)
    const int s    = blockIdx.x & 15;   // column split (32 h-cols)
    const int b0   = g * 16;
    const int lane = tid & 31;
    const int lr   = lane & 15;
    const int half = lane >> 4;
    const int wv   = tid >> 5;          // wave 0..7 -> gate q = wv>>1, coltile c = wv&1
    const int q    = wv >> 1;
    const int ccol = (wv & 1) * 16 + lr;
    const int ntg  = q * 32 + s * 2 + (wv & 1);   // global ntile of this wave

    unsigned short* hx = Hex + (size_t)g * HEXSTRIDE;

    // Load this block's W_hh slice (8 ntiles x 16KB) into LDS once.
    for (int i8 = tid; i8 < 8192; i8 += 256) {
        int n   = i8 >> 10;
        int off = (i8 & 1023) * 8;
        int nsrc = (n >> 1) * 32 + s * 2 + (n & 1);
        *(v8u*)&w_lds[n * 8192 + off] = *(const v8u*)&Wp[(size_t)nsrc * 8192 + off];
    }

    // gate-phase ownership: thread -> rows {r, r+8}, one column
    const int grow = tid >> 5;   // 0..7
    const int gcol = tid & 31;   // 0..31
    float c_st[2] = {0.f, 0.f};

    __syncthreads();

    for (int t = 0; t < T_; ++t) {
        // stage h (bf16) from the exchange buffer into LDS
        for (int i8 = tid; i8 < 1024; i8 += 256)
            *(v8u*)&h_lds[i8 * 8] = *(const v8u*)&hx[i8 * 8];
        __syncthreads();

        // GEMM: one 16x16 gate tile per wave, K = 512, everything from LDS
        v8f acc = {};
#pragma unroll
        for (int kb = 0; kb < 16; ++kb) {
            const int k0 = kb * 32;
            v16u af = cat16(*(const v8u*)&h_lds[lr * H_ + k0 + half * 16],
                            *(const v8u*)&h_lds[lr * H_ + k0 + half * 16 + 8]);
            const unsigned short* bp = &w_lds[((wv * 16 + kb) * 32 + lane) * 16];
            v16u bf = cat16(*(const v8u*)bp, *(const v8u*)(bp + 8));
            acc = wmma_bf16(af, bf, acc);
        }

        // add input projection and expose tile through LDS for gate fusion
        const size_t xbase = ((size_t)(b0 + half * 8) * T_ + t) * (size_t)G4 + ntg * 16 + lr;
#pragma unroll
        for (int v = 0; v < 8; ++v) {
            const size_t xo = xbase + (size_t)v * ((size_t)T_ * G4);
            __builtin_prefetch(&XG[xo + G4], 0, 1);   // next timestep
            g_lds[q][half * 8 + v][ccol] = acc[v] + XG[xo];
        }
        __syncthreads();

        // fused LSTM cell: each thread owns (row, col) and (row+8, col)
#pragma unroll
        for (int rr = 0; rr < 2; ++rr) {
            const int row = grow + rr * 8;
            const float xi = g_lds[0][row][gcol];
            const float xf = g_lds[1][row][gcol];
            const float xg = g_lds[2][row][gcol];
            const float xo = g_lds[3][row][gcol];
            const float ig = sigmoidf_(xi);
            const float fg = sigmoidf_(xf);
            const float gg = tanhf(xg);
            const float og = sigmoidf_(xo);
            const float c  = fg * c_st[rr] + ig * gg;
            c_st[rr] = c;
            const float h = og * tanhf(c);
            const int col = s * WGCOLS + gcol;
            hx[row * H_ + col] = f2bf(h);
            if (HSeq)  HSeq[((size_t)(b0 + row) * T_ + t) * H_ + col] = h;
            if (HLast && t == T_ - 1) HLast[(size_t)(b0 + row) * H_ + col] = h;
        }
        __threadfence();      // make this block's h stores device-visible
        __syncthreads();

        // device-scope barrier across the 16 blocks of this batch group
        if (tid == 0) {
            __hip_atomic_fetch_add(&Bar[g], 1u, __ATOMIC_RELEASE, __HIP_MEMORY_SCOPE_AGENT);
            const unsigned int target = 16u * (unsigned int)(t + 1);
            while (__hip_atomic_load(&Bar[g], __ATOMIC_ACQUIRE, __HIP_MEMORY_SCOPE_AGENT) < target)
                __builtin_amdgcn_s_sleep(1);
        }
        __syncthreads();
        // drop any stale h-exchange lines from WGP$ before next step's reads
        asm volatile("global_inv scope:SCOPE_DEV\n\ts_wait_loadcnt 0x0" ::: "memory");
    }
}

// out[b] = dot(h_last[b,:], Wl) + bl
__global__ void final_linear(const float* __restrict__ Hl, const float* __restrict__ Wl,
                             const float* __restrict__ bl, float* __restrict__ out) {
    int b = blockIdx.x * blockDim.x + threadIdx.x;
    if (b >= B_) return;
    float sacc = 0.f;
    for (int k = 0; k < H_; ++k) sacc += Hl[(size_t)b * H_ + k] * Wl[k];
    out[b] = sacc + bl[0];
}

extern "C" void kernel_launch(void* const* d_in, const int* in_sizes, int n_in,
                              void* d_out, int out_size, void* d_ws, size_t ws_size,
                              hipStream_t stream) {
    const float* x    = (const float*)d_in[0];
    const float* Wih0 = (const float*)d_in[1];
    const float* Whh0 = (const float*)d_in[2];
    const float* bih0 = (const float*)d_in[3];
    const float* bhh0 = (const float*)d_in[4];
    const float* Wih1 = (const float*)d_in[5];
    const float* Whh1 = (const float*)d_in[6];
    const float* bih1 = (const float*)d_in[7];
    const float* bhh1 = (const float*)d_in[8];
    const float* Wl   = (const float*)d_in[9];
    const float* bl   = (const float*)d_in[10];

    // Workspace layout
    float* xg    = (float*)d_ws;                               // B*T*4H f32 (reused by both layers)
    float* hseq  = xg + (size_t)B_ * T_ * G4;                  // B*T*H f32
    float* hlast = hseq + (size_t)B_ * T_ * H_;                // B*H f32
    unsigned short* wih0p = (unsigned short*)(hlast + (size_t)B_ * H_);
    unsigned short* whh0p = wih0p + (size_t)G4 * IN_;
    unsigned short* wih1p = whh0p + (size_t)G4 * H_;
    unsigned short* whh1p = wih1p + (size_t)G4 * H_;
    unsigned short* hexb  = whh1p + (size_t)G4 * H_;           // 4*16*512 bf16 h-exchange
    unsigned int*   bar   = (unsigned int*)(hexb + 4 * HEXSTRIDE);

    // Pack weights into WMMA B-fragment bf16 layout.
    pack_w<<<(G4 * IN_) / 256, 256, 0, stream>>>(Wih0, wih0p, G4, IN_);
    pack_w<<<(G4 * H_ ) / 256, 256, 0, stream>>>(Whh0, whh0p, G4, H_);
    pack_w<<<(G4 * H_ ) / 256, 256, 0, stream>>>(Wih1, wih1p, G4, H_);
    pack_w<<<(G4 * H_ ) / 256, 256, 0, stream>>>(Whh1, whh1p, G4, H_);

    // Layer 0
    gemm_xg<<<(B_ * T_) / 16, 256, 0, stream>>>(x, wih0p, bih0, bhh0, xg, IN_);
    scan_init<<<32, 256, 0, stream>>>(hexb, bar);
    lstm_scan<<<64, 256, 0, stream>>>(xg, whh0p, hseq, nullptr, hexb, bar);
    // Layer 1
    gemm_xg<<<(B_ * T_) / 16, 256, 0, stream>>>(hseq, wih1p, bih1, bhh1, xg, H_);
    scan_init<<<32, 256, 0, stream>>>(hexb, bar);
    lstm_scan<<<64, 256, 0, stream>>>(xg, whh1p, nullptr, hlast, hexb, bar);
    // Head
    final_linear<<<1, 64, 0, stream>>>(hlast, Wl, bl, (float*)d_out);
}
